// TBStars2MoESparseBlock_18614388261194
// MI455X (gfx1250) — matrix operand order repaired and checked by reading, not compiled, not run on hardware
//
#include <hip/hip_runtime.h>
#include <hip/hip_bf16.h>

typedef __bf16 bf16_t;
typedef __attribute__((ext_vector_type(16))) __bf16 v16bf;
typedef __attribute__((ext_vector_type(8)))  __bf16 v8bf;
typedef __attribute__((ext_vector_type(8)))  float  v8f;

#define HIDDEN 2048
#define FFN    1408
#define NEXP   8
#define SEQ    4096
#define BM     64
#define BN     64
#define BK     32
#define AS     40   // LDS row stride in bf16 elements (32 data + 8 pad); 80B rows, 16B aligned

// ---------------- helpers ----------------

__device__ __forceinline__ v16bf mk_frag(const bf16_t* p0, const bf16_t* p1) {
  v8bf lo = *(const v8bf*)p0;
  v8bf hi = *(const v8bf*)p1;
  v16bf r;
#pragma unroll
  for (int i = 0; i < 8; ++i) { r[i] = lo[i]; r[i + 8] = hi[i]; }
  return r;
}

__device__ __forceinline__ v8bf cvt8(float4 a, float4 b) {
  v8bf p;
  p[0] = (bf16_t)a.x; p[1] = (bf16_t)a.y; p[2] = (bf16_t)a.z; p[3] = (bf16_t)a.w;
  p[4] = (bf16_t)b.x; p[5] = (bf16_t)b.y; p[6] = (bf16_t)b.z; p[7] = (bf16_t)b.w;
  return p;
}

// ---------------- router ----------------
__global__ __launch_bounds__(256) void router_kernel(
    const float* __restrict__ hs, const float* __restrict__ gw,
    float* __restrict__ logits_out, int* __restrict__ counts,
    int* __restrict__ row_token, float* __restrict__ row_weight)
{
  const int t   = blockIdx.x;
  const int tid = threadIdx.x;
  float acc[NEXP];
#pragma unroll
  for (int e = 0; e < NEXP; ++e) acc[e] = 0.0f;

  const float* x = hs + (size_t)t * HIDDEN;
  for (int h = tid; h < HIDDEN; h += 256) {
    float xv = x[h];
    const float* g = gw + (size_t)h * NEXP;
#pragma unroll
    for (int e = 0; e < NEXP; ++e) acc[e] += xv * g[e];
  }

  __shared__ float red[NEXP][256];
#pragma unroll
  for (int e = 0; e < NEXP; ++e) red[e][tid] = acc[e];

  for (int s = 128; s > 0; s >>= 1) {
    __syncthreads();
    if (tid < s) {
#pragma unroll
      for (int e = 0; e < NEXP; ++e) red[e][tid] += red[e][tid + s];
    }
  }
  __syncthreads();

  if (tid == 0) {
    float l[NEXP];
#pragma unroll
    for (int e = 0; e < NEXP; ++e) { l[e] = red[e][0]; logits_out[(size_t)t * NEXP + e] = l[e]; }
    int i0 = 0;
#pragma unroll
    for (int e = 1; e < NEXP; ++e) if (l[e] > l[i0]) i0 = e;
    int i1 = (i0 == 0) ? 1 : 0;
#pragma unroll
    for (int e = 0; e < NEXP; ++e) if (e != i0 && l[e] > l[i1]) i1 = e;
    float w0 = 1.0f / (1.0f + __expf(l[i1] - l[i0]));
    float w1 = 1.0f - w0;
    int s0 = atomicAdd(&counts[i0], 1);
    row_token[i0 * SEQ + s0]  = t;
    row_weight[i0 * SEQ + s0] = w0;
    int s1 = atomicAdd(&counts[i1], 1);
    row_token[i1 * SEQ + s1]  = t;
    row_weight[i1 * SEQ + s1] = w1;
  }
}

__global__ void prefix_kernel(const int* __restrict__ counts, int* __restrict__ offs) {
  if (threadIdx.x == 0) {
    int s = 0;
#pragma unroll
    for (int e = 0; e < NEXP; ++e) { offs[e] = s; s += counts[e]; }
    offs[NEXP] = s;
  }
}

// ---------------- expert GEMM 1: act = silu(X @ W1g^T) * (X @ W1u^T) ----------------
// Double-buffered LDS, one barrier per K-step.
__global__ __launch_bounds__(256) void moe_gemm1(
    const float* __restrict__ hs, const float* __restrict__ w1,
    const int* __restrict__ counts, const int* __restrict__ offs,
    const int* __restrict__ row_token, bf16_t* __restrict__ act)
{
  const int e   = blockIdx.z;
  const int cnt = counts[e];
  const int m0  = blockIdx.y * BM;
  if (m0 >= cnt) return;
  const int f0     = blockIdx.x * BN;
  const int rowoff = offs[e];

  __shared__ bf16_t ldsA [2][BM * AS];
  __shared__ bf16_t ldsBg[2][BN * AS];
  __shared__ bf16_t ldsBu[2][BN * AS];

  const int tid  = threadIdx.x;
  const int lane = tid & 31;
  const int wave = tid >> 5;
  const int lg   = lane >> 4;
  const int ln   = lane & 15;
  const int rb   = (wave & 3) * 16;    // 4 row blocks of 16
  const int cb   = (wave >> 2) * 32;   // 2 col blocks of 32

  const int lrow = tid >> 2;           // 0..63
  const int lk   = (tid & 3) * 8;      // 8-elem K chunk
  const int sidx = lrow * AS + lk;

  const int slot = m0 + lrow;
  const int tok  = (slot < cnt) ? row_token[e * SEQ + slot] : 0;   // clamped rows never stored
  const float* aSrc = hs + (size_t)tok * HIDDEN + lk;
  const float* gSrc = w1 + ((size_t)e * (2 * FFN) + (size_t)(f0 + lrow)) * HIDDEN + lk;
  const float* uSrc = gSrc + (size_t)FFN * HIDDEN;

  v8f accg0 = {}, accg1 = {}, accu0 = {}, accu1 = {};
  float4 pa0, pa1, pg0, pg1, pu0, pu1;

#define LDG1(K) do {                                                   \
    pa0 = *(const float4*)(aSrc + (K)); pa1 = *(const float4*)(aSrc + (K) + 4); \
    pg0 = *(const float4*)(gSrc + (K)); pg1 = *(const float4*)(gSrc + (K) + 4); \
    pu0 = *(const float4*)(uSrc + (K)); pu1 = *(const float4*)(uSrc + (K) + 4); } while (0)
#define STS1(B) do {                                                   \
    *(v8bf*)&ldsA [B][sidx] = cvt8(pa0, pa1);                          \
    *(v8bf*)&ldsBg[B][sidx] = cvt8(pg0, pg1);                          \
    *(v8bf*)&ldsBu[B][sidx] = cvt8(pu0, pu1); } while (0)

  LDG1(0);
  STS1(0);
  __syncthreads();

  int cur = 0;
  for (int k0 = 0; k0 < HIDDEN; k0 += BK) {
    const bool hasNext = (k0 + BK) < HIDDEN;
    if (hasNext) LDG1(k0 + BK);

    const bf16_t* ap  = &ldsA [cur][(rb + ln) * AS + lg * 16];
    v16bf a  = mk_frag(ap, ap + 8);
    const bf16_t* bg0 = &ldsBg[cur][(cb + ln) * AS + lg * 16];
    const bf16_t* bg1 = &ldsBg[cur][(cb + 16 + ln) * AS + lg * 16];
    const bf16_t* bu0 = &ldsBu[cur][(cb + ln) * AS + lg * 16];
    const bf16_t* bu1 = &ldsBu[cur][(cb + 16 + ln) * AS + lg * 16];
    v16bf fg0 = mk_frag(bg0, bg0 + 8);
    v16bf fg1 = mk_frag(bg1, bg1 + 8);
    v16bf fu0 = mk_frag(bu0, bu0 + 8);
    v16bf fu1 = mk_frag(bu1, bu1 + 8);

    accg0 = __builtin_amdgcn_wmma_f32_16x16x32_bf16(false, a, false, fg0, (short)0, accg0, false, false);
    accg1 = __builtin_amdgcn_wmma_f32_16x16x32_bf16(false, a, false, fg1, (short)0, accg1, false, false);
    accu0 = __builtin_amdgcn_wmma_f32_16x16x32_bf16(false, a, false, fu0, (short)0, accu0, false, false);
    accu1 = __builtin_amdgcn_wmma_f32_16x16x32_bf16(false, a, false, fu1, (short)0, accu1, false, false);

    if (hasNext) STS1(cur ^ 1);
    __syncthreads();
    cur ^= 1;
  }

#pragma unroll
  for (int nb = 0; nb < 2; ++nb) {
    v8f g = nb ? accg1 : accg0;
    v8f u = nb ? accu1 : accu0;
#pragma unroll
    for (int v = 0; v < 8; ++v) {
      int row = rb + v + lg * 8;
      int s   = m0 + row;
      if (s < cnt) {
        int col  = f0 + cb + nb * 16 + ln;
        float gv = g[v];
        float av = (gv / (1.0f + __expf(-gv))) * u[v];
        act[(size_t)(rowoff + s) * FFN + col] = (bf16_t)av;
      }
    }
  }
#undef LDG1
#undef STS1
}

// ---------------- expert GEMM 2: out[token] += weight * (act @ W2^T) ----------------
// Double-buffered LDS; A-tile is already bf16 (pure 16B copies), B converted f32->bf16.
__global__ __launch_bounds__(256) void moe_gemm2(
    const bf16_t* __restrict__ act, const float* __restrict__ w2,
    const int* __restrict__ counts, const int* __restrict__ offs,
    const int* __restrict__ row_token, const float* __restrict__ row_weight,
    float* __restrict__ out)
{
  const int e   = blockIdx.z;
  const int cnt = counts[e];
  const int m0  = blockIdx.y * BM;
  if (m0 >= cnt) return;
  const int h0     = blockIdx.x * BN;
  const int rowoff = offs[e];

  __shared__ bf16_t ldsA[2][BM * AS];
  __shared__ bf16_t ldsB[2][BN * AS];

  const int tid  = threadIdx.x;
  const int lane = tid & 31;
  const int wave = tid >> 5;
  const int lg   = lane >> 4;
  const int ln   = lane & 15;
  const int rb   = (wave & 3) * 16;
  const int cb   = (wave >> 2) * 32;

  const int lrow = tid >> 2;
  const int lk   = (tid & 3) * 8;
  const int sidx = lrow * AS + lk;

  const int slot = m0 + lrow;
  const int arow = (slot < cnt) ? (rowoff + slot) : rowoff;   // clamp inside act buffer
  const bf16_t* aSrc = act + (size_t)arow * FFN + lk;
  const float*  bSrc = w2 + ((size_t)e * HIDDEN + (size_t)(h0 + lrow)) * FFN + lk;

  v8f acc0 = {}, acc1 = {};
  v8bf  pa;
  float4 pb0, pb1;

#define LDG2(K) do {                                                   \
    pa  = *(const v8bf*)(aSrc + (K));                                  \
    pb0 = *(const float4*)(bSrc + (K)); pb1 = *(const float4*)(bSrc + (K) + 4); } while (0)
#define STS2(B) do {                                                   \
    *(v8bf*)&ldsA[B][sidx] = pa;                                       \
    *(v8bf*)&ldsB[B][sidx] = cvt8(pb0, pb1); } while (0)

  LDG2(0);
  STS2(0);
  __syncthreads();

  int cur = 0;
  for (int k0 = 0; k0 < FFN; k0 += BK) {
    const bool hasNext = (k0 + BK) < FFN;
    if (hasNext) LDG2(k0 + BK);

    const bf16_t* ap  = &ldsA[cur][(rb + ln) * AS + lg * 16];
    v16bf a = mk_frag(ap, ap + 8);
    const bf16_t* b0p = &ldsB[cur][(cb + ln) * AS + lg * 16];
    const bf16_t* b1p = &ldsB[cur][(cb + 16 + ln) * AS + lg * 16];
    v16bf fb0 = mk_frag(b0p, b0p + 8);
    v16bf fb1 = mk_frag(b1p, b1p + 8);

    acc0 = __builtin_amdgcn_wmma_f32_16x16x32_bf16(false, a, false, fb0, (short)0, acc0, false, false);
    acc1 = __builtin_amdgcn_wmma_f32_16x16x32_bf16(false, a, false, fb1, (short)0, acc1, false, false);

    if (hasNext) STS2(cur ^ 1);
    __syncthreads();
    cur ^= 1;
  }

#pragma unroll
  for (int nb = 0; nb < 2; ++nb) {
    v8f c = nb ? acc1 : acc0;
#pragma unroll
    for (int v = 0; v < 8; ++v) {
      int row = rb + v + lg * 8;
      int s   = m0 + row;
      if (s < cnt) {
        int tok = row_token[e * SEQ + s];
        float w = row_weight[e * SEQ + s];
        int col = h0 + cb + nb * 16 + ln;
        // exactly TOP_K=2 commutative f32 adds per element -> bitwise deterministic
        unsafeAtomicAdd(&out[(size_t)tok * HIDDEN + col], w * c[v]);
      }
    }
  }
#undef LDG2
#undef STS2
}

// ---------------- launch ----------------
extern "C" void kernel_launch(void* const* d_in, const int* in_sizes, int n_in,
                              void* d_out, int out_size, void* d_ws, size_t ws_size,
                              hipStream_t stream) {
  (void)in_sizes; (void)n_in; (void)out_size; (void)ws_size;
  const float* hs = (const float*)d_in[0];   // [SEQ, HIDDEN]
  const float* gw = (const float*)d_in[1];   // [HIDDEN, NEXP]
  const float* w1 = (const float*)d_in[2];   // [NEXP, 2*FFN, HIDDEN]
  const float* w2 = (const float*)d_in[3];   // [NEXP, HIDDEN, FFN]
  float* out = (float*)d_out;                // [SEQ*HIDDEN] ++ [SEQ*NEXP]
  float* logits_out = out + (size_t)SEQ * HIDDEN;

  char* ws = (char*)d_ws;
  int*    counts     = (int*)(ws + 0);                           // 8 ints
  int*    offs       = (int*)(ws + 256);                         // 9 ints
  int*    row_token  = (int*)(ws + 512);                         // [NEXP*SEQ]
  float*  row_weight = (float*)(ws + 512 + NEXP * SEQ * 4);      // [NEXP*SEQ]
  bf16_t* act        = (bf16_t*)(ws + 512 + 2 * NEXP * SEQ * 4); // [2*SEQ, FFN] bf16 (~23 MB)

  hipError_t err;
  err = hipMemsetAsync(counts, 0, 256, stream); (void)err;
  err = hipMemsetAsync(out, 0, (size_t)SEQ * HIDDEN * sizeof(float), stream); (void)err;

  router_kernel<<<dim3(SEQ), dim3(256), 0, stream>>>(hs, gw, logits_out, counts, row_token, row_weight);
  prefix_kernel<<<dim3(1), dim3(32), 0, stream>>>(counts, offs);

  dim3 g1(FFN / BN, SEQ / BM, NEXP);      // (22, 64, 8); M-tiles beyond count early-exit
  moe_gemm1<<<g1, dim3(256), 0, stream>>>(hs, w1, counts, offs, row_token, act);

  dim3 g2(HIDDEN / BN, SEQ / BM, NEXP);   // (32, 64, 8)
  moe_gemm2<<<g2, dim3(256), 0, stream>>>(act, w2, counts, offs, row_token, row_weight, out);
}
